// Attention_35888746726185
// MI455X (gfx1250) — matrix operand order repaired and checked by reading, not compiled, not run on hardware
//
#include <hip/hip_runtime.h>
#include <hip/hip_bf16.h>
#include <math.h>

// Problem constants (from the reference)
#define BSZ   4
#define DIMC  128
#define QKVC  384          // 3*DIM
#define HEADS 8
#define CH    16           // DIM/HEADS
#define HDIM  256
#define WDIM  256
#define HW    65536
#define GC    16           // DIM/8
#define IDCH  80           // DIM - 3*GC
#define EPSN  1e-12f

// Workspace (bytes): qkv bf16 (192MB) + qkvc bf16 (192MB) + sumsq (4KB) +
// G (32KB) + Wc bf16 (128KB)  ~= 403 MB total.

typedef __attribute__((ext_vector_type(16))) __bf16 v16bf;
typedef __attribute__((ext_vector_type(8)))  float  v8f;
typedef __attribute__((ext_vector_type(2)))  float  v2f;
typedef __attribute__((ext_vector_type(4)))  unsigned int v4u;
typedef __attribute__((ext_vector_type(8)))  int    v8i;
typedef __attribute__((ext_vector_type(4)))  int    v4i;

union Frag16 { v16bf v; unsigned short s[16]; unsigned int d[8]; };

static __device__ __forceinline__ unsigned short f2bf(float f) {
  __bf16 h = (__bf16)f;                     // native v_cvt (RNE)
  return __builtin_bit_cast(unsigned short, h);
}
static __device__ __forceinline__ float bf2f(unsigned short u) {
  unsigned int x = ((unsigned int)u) << 16;
  return __builtin_bit_cast(float, x);
}

// ---------------------------------------------------------------------------
// TDM: async DMA of a 2D tile (16 rows x 512 bf16, row stride HW elements)
// from global into LDS with padding: +4 dwords after every 256 dwords, so the
// LDS row stride is 520 bf16 -> conflict-free ds reads in the WMMA stage.
// Tracked by TENSORcnt; drained with s_wait_tensorcnt.
// ---------------------------------------------------------------------------
static __device__ __forceinline__ void tdm_load_tile(const unsigned short* gsrc,
                                                     void* lds_dst) {
  unsigned long long ga = (unsigned long long)(size_t)gsrc;
  unsigned int lds_addr = (unsigned int)(size_t)lds_dst;  // LDS aperture: low 32 bits
  v4u g0;
  g0[0] = 1u;                                          // count=1 (valid user D#)
  g0[1] = lds_addr;                                    // lds_addr [63:32]
  g0[2] = (unsigned int)ga;                            // global_addr low  [95:64]
  g0[3] = ((unsigned int)(ga >> 32) & 0x01FFFFFFu)     // global_addr high [120:96]
          | (2u << 30);                                // type=2 ("image") [127:126]
  v8i g1;
  g1[0] = (int)((1u << 16)      // data_size = 1 -> 2 bytes
              | (1u << 20)      // pad_enable
              | (7u << 22)      // pad_interval: code 7 = every 256 dwords (1024B = 1 row)
              | (3u << 25));    // pad_amount:   code 3 = 4 dwords (8 bf16)
  g1[1] = 0;                    // atomic_barrier_addr=0; tensor_dim0 lo16 (65536 -> 0)
  g1[2] = (int)((16u << 16) | 1u); // tensor_dim0 hi16 (0x1 -> 65536), tensor_dim1 = 16
  g1[3] = (int)(512u << 16);    // tensor_dim1 hi = 0, tile_dim0 = 512 elements
  g1[4] = 16;                   // tile_dim1 = 16 rows, tile_dim2 = 0
  g1[5] = 65536;                // tensor_dim0_stride low 32 = HW elements
  g1[6] = 0;                    // stride hi16, tensor_dim1_stride lo16
  g1[7] = 0;
  v4i z4 = {0, 0, 0, 0};
#if defined(__clang_major__) && (__clang_major__ >= 23)
  v8i z8 = {0, 0, 0, 0, 0, 0, 0, 0};
  __builtin_amdgcn_tensor_load_to_lds(g0, g1, z4, z4, z8, 0);
#else
  __builtin_amdgcn_tensor_load_to_lds(g0, g1, z4, z4, 0);
#endif
}

// ---------------------------------------------------------------------------
// K0: zero sumsq + gram accumulators (they are atomically accumulated)
// ---------------------------------------------------------------------------
__global__ __launch_bounds__(256) void k0_zero(float* __restrict__ p, int n) {
  int i = blockIdx.x * 256 + threadIdx.x;
  if (i < n) p[i] = 0.f;
}

// ---------------------------------------------------------------------------
// K1: qkv[b,o,n] = sum_c x[b,c,n] * qkv_w[o,c]   (bf16 WMMA 16x16x32)
// One wave = one 16-pixel tile, loops all 24 o-tiles reusing the x fragments.
// qkv_w staged once per block into padded LDS (bank-conflict-avoiding stride).
// ---------------------------------------------------------------------------
__global__ __launch_bounds__(256) void k1_qkv_gemm(const float* __restrict__ x,
                                                   const float* __restrict__ qkv_w,
                                                   unsigned short* __restrict__ qkv) {
  extern __shared__ unsigned short aLds[];            // 384 rows * 130 bf16 (padded)
  const int tid = threadIdx.x;
  for (int i = tid; i < QKVC * DIMC; i += 256) {
    int o = i >> 7, c = i & 127;
    aLds[o * 130 + c] = f2bf(qkv_w[i]);
  }
  __syncthreads();

  const int wave = tid >> 5, lane = tid & 31;
  const int col = lane & 15, half = lane >> 4;
  const size_t tile = (size_t)blockIdx.x * 8 + wave;  // 16-pixel tile id
  const size_t pix0 = tile * 16;
  const int b = (int)(pix0 >> 16);
  const size_t sp = pix0 & 65535;
  const float* xb = x + (size_t)b * DIMC * HW;
  const size_t myPix = sp + col;

  // B fragments (32x16, 16-bit): lanes 0-15 hold K=0..15, lanes 16-31 K=16..31
  Frag16 bfrag[4];
#pragma unroll
  for (int ks = 0; ks < 4; ++ks) {
    __builtin_prefetch(xb + (size_t)(32 * ks + 16 * half) * HW + myPix, 0, 1);
#pragma unroll
    for (int e = 0; e < 16; ++e) {
      int c = 32 * ks + 16 * half + e;
      bfrag[ks].s[e] = f2bf(xb[(size_t)c * HW + myPix]);
    }
  }

  for (int ot = 0; ot < QKVC / 16; ++ot) {
    v8f acc = {0.f, 0.f, 0.f, 0.f, 0.f, 0.f, 0.f, 0.f};
#pragma unroll
    for (int ks = 0; ks < 4; ++ks) {
      // A fragment (16x32, 16-bit): lane row M=col; K 0..15 (+16 for upper half)
      Frag16 af;
      const unsigned short* arow = aLds + (ot * 16 + col) * 130 + 32 * ks + (half ? 16 : 0);
#pragma unroll
      for (int v = 0; v < 8; ++v) {
        int c0 = 2 * v;
        af.d[v] = *(const unsigned int*)(arow + c0);   // ds_load_b32
      }
      acc = __builtin_amdgcn_wmma_f32_16x16x32_bf16(false, af.v, false, bfrag[ks].v,
                                                    (short)0, acc, false, false);
    }
#pragma unroll
    for (int r = 0; r < 8; ++r) {
      int orow = ot * 16 + r + 8 * half;               // C layout: VGPR r -> M=r (+8 upper half)
      qkv[((size_t)b * QKVC + orow) * HW + sp + col] = f2bf(acc[r]);
    }
  }
}

// ---------------------------------------------------------------------------
// K2: inception depthwise convs (identity / 3x3 / 1x11 / 11x1) + per-channel
// sum-of-squares for q,k (folded normalization). One block = one (b,c,row).
// ---------------------------------------------------------------------------
struct ConvW {
  const float* hw_w[3]; const float* hw_b[3];
  const float* w_w[3];  const float* w_b[3];
  const float* h_w[3];  const float* h_b[3];
};

static __device__ __forceinline__ float ldpix(const unsigned short* ch, int h, int w) {
  if ((unsigned)h >= (unsigned)HDIM || (unsigned)w >= (unsigned)WDIM) return 0.f;
  return bf2f(ch[h * WDIM + w]);
}

__global__ __launch_bounds__(256) void k2_inception(const unsigned short* __restrict__ qkv,
                                                    unsigned short* __restrict__ qkvc,
                                                    float* __restrict__ sumsq,
                                                    ConvW cw) {
  const int w = threadIdx.x;
  const int h = blockIdx.x;
  const int c = blockIdx.y;
  const int b = blockIdx.z;
  const int p = c >> 7;          // 0=q 1=k 2=v
  const int cl = c & 127;
  const unsigned short* src = qkv + ((size_t)b * QKVC + c) * HW;

  float y;
  if (cl < IDCH) {
    y = bf2f(src[h * WDIM + w]);
  } else if (cl < IDCH + GC) {
    int gi = cl - IDCH;
    const float* wg = cw.hw_w[p] + gi * 9;
    float s = cw.hw_b[p][gi];
#pragma unroll
    for (int dy = -1; dy <= 1; ++dy)
#pragma unroll
      for (int dx = -1; dx <= 1; ++dx)
        s += wg[(dy + 1) * 3 + (dx + 1)] * ldpix(src, h + dy, w + dx);
    y = s;
  } else if (cl < IDCH + 2 * GC) {
    int gi = cl - (IDCH + GC);
    const float* wg = cw.w_w[p] + gi * 11;
    float s = cw.w_b[p][gi];
#pragma unroll
    for (int dx = -5; dx <= 5; ++dx)
      s += wg[dx + 5] * ldpix(src, h, w + dx);
    y = s;
  } else {
    int gi = cl - (IDCH + 2 * GC);
    const float* wg = cw.h_w[p] + gi * 11;
    float s = cw.h_b[p][gi];
#pragma unroll
    for (int dy = -5; dy <= 5; ++dy)
      s += wg[dy + 5] * ldpix(src, h + dy, w);
    y = s;
  }
  qkvc[((size_t)b * QKVC + c) * HW + h * WDIM + w] = f2bf(y);

  // sum of squares for q,k channels (block tree-reduce + one global f32 atomic)
  __shared__ float red[256];
  red[threadIdx.x] = (p < 2) ? y * y : 0.f;
  __syncthreads();
  for (int s2 = 128; s2 > 0; s2 >>= 1) {
    if (threadIdx.x < s2) red[threadIdx.x] += red[threadIdx.x + s2];
    __syncthreads();
  }
  if (threadIdx.x == 0 && p < 2)
    atomicAdd(&sumsq[b * 256 + p * 128 + cl], red[0]);
}

// ---------------------------------------------------------------------------
// K3: raw gram G[b,h][c,d] = sum_n q[c,n]*k[d,n] via v_wmma_f32_16x16x4_f32.
// TDM double-buffered: wave 0 issues tensor_load_to_lds for the next stage's
// q/k tiles while all 8 waves run WMMAs on the current buffer; stage handoff
// is s_wait_tensorcnt (issuing wave) + workgroup barrier. LDS rows are padded
// by the TDM descriptor to 520 bf16 -> conflict-free ds reads.
// Split-K over 8 blocks per (b,h); LDS ds_add_f32 merge then global atomics.
// NOTE: LDS buffer pointers are derived with integer offsets only (no pointer
// arrays) -- constant aggregates of LDS addrspacecasts break ld.lld.
// ---------------------------------------------------------------------------
#define K3_TILE_US (16 * 520)                 // ushorts per staged matrix
__global__ __launch_bounds__(256) void k3_gram(const unsigned short* __restrict__ qkvc,
                                               float* __restrict__ G) {
  extern __shared__ unsigned char smem3[];
  unsigned short* sbase = (unsigned short*)smem3;       // [buf0 q][buf0 k][buf1 q][buf1 k]
  float* gsh = (float*)(sbase + 4 * K3_TILE_US);        // 256 floats
  const int tid = threadIdx.x;
  gsh[tid] = 0.f;

  const int bh = blockIdx.y;
  const int b = bh >> 3, h = bh & 7;
  const unsigned short* qb = qkvc + ((size_t)b * QKVC + h * CH) * HW;
  const unsigned short* kb = qkvc + ((size_t)b * QKVC + DIMC + h * CH) * HW;
  const int wave = tid >> 5, lane = tid & 31;
  const int col = lane & 15, half = lane >> 4;
  const size_t chunk0 = (size_t)blockIdx.x * 8192;

  if (wave == 0) {                             // prime the pipeline (stage 0)
    tdm_load_tile(qb + chunk0, sbase);
    tdm_load_tile(kb + chunk0, sbase + K3_TILE_US);
  }

  v8f acc = {0.f, 0.f, 0.f, 0.f, 0.f, 0.f, 0.f, 0.f};
  for (int st = 0; st < 16; ++st) {
    const int cur = st & 1;
    if (wave == 0) {
      if (st < 15) {                           // issue next stage into other buffer
        size_t n1 = chunk0 + (size_t)(st + 1) * 512;
        unsigned short* nq = sbase + (size_t)(cur ^ 1) * (2 * K3_TILE_US);
        tdm_load_tile(qb + n1, nq);
        tdm_load_tile(kb + n1, nq + K3_TILE_US);
        __builtin_amdgcn_s_wait_tensorcnt(2);  // oldest 2 (current stage) done
      } else {
        __builtin_amdgcn_s_wait_tensorcnt(0);
      }
    }
    __syncthreads();                           // publish DMA'd tiles to all waves
    const unsigned short* qs  = sbase + (size_t)cur * (2 * K3_TILE_US);
    const unsigned short* ksm = qs + K3_TILE_US;
#pragma unroll
    for (int j = 0; j < 16; ++j) {             // this wave covers 64 n per stage
      int nl = wave * 64 + j * 4 + 2 * half;   // A: VGPR0=K(2*half), VGPR1=K+1
      unsigned int qd = *(const unsigned int*)(qs  + col * 520 + nl);
      unsigned int kd = *(const unsigned int*)(ksm + col * 520 + nl);
      v2f a, bb;
      a.x  = bf2f((unsigned short)(qd & 0xFFFFu));
      a.y  = bf2f((unsigned short)(qd >> 16));
      bb.x = bf2f((unsigned short)(kd & 0xFFFFu));
      bb.y = bf2f((unsigned short)(kd >> 16));
      acc = __builtin_amdgcn_wmma_f32_16x16x4_f32(false, a, false, bb,
                                                  (short)0, acc, false, false);
    }
    __syncthreads();                           // buffer may be re-filled next iter
  }
#pragma unroll
  for (int r = 0; r < 8; ++r)                  // merge 8 waves via LDS f32 atomics
    atomicAdd(&gsh[(r + 8 * half) * 16 + col], acc[r]);
  __syncthreads();
  atomicAdd(&G[(size_t)bh * 256 + tid], gsh[tid]);     // split-K global merge
}

// ---------------------------------------------------------------------------
// K4: normalize gram by ||q_c||*||k_d||, * temperature, softmax over d, then
// fold proj_w through the block-diagonal attn: Wc[b][o][h*16+d].
// ---------------------------------------------------------------------------
__global__ __launch_bounds__(256) void k4_attn(const float* __restrict__ G,
                                               const float* __restrict__ sumsq,
                                               const float* __restrict__ temperature,
                                               const float* __restrict__ proj_w,
                                               unsigned short* __restrict__ Wc) {
  __shared__ float attn[HEADS][CH][CH];
  const int b = blockIdx.x;
  const int tid = threadIdx.x;
  if (tid < 128) {
    int h = tid >> 4, c = tid & 15;
    float qn = fmaxf(sqrtf(sumsq[b * 256 + h * CH + c]), EPSN);
    float t = temperature[h];
    float row[16];
    float mx = -3.4e38f;
#pragma unroll
    for (int d = 0; d < 16; ++d) {
      float kn = fmaxf(sqrtf(sumsq[b * 256 + 128 + h * CH + d]), EPSN);
      float s = G[((size_t)(b * HEADS + h)) * 256 + c * 16 + d] / (qn * kn) * t;
      row[d] = s;
      mx = fmaxf(mx, s);
    }
    float sum = 0.f;
#pragma unroll
    for (int d = 0; d < 16; ++d) { row[d] = __expf(row[d] - mx); sum += row[d]; }
    float inv = 1.f / sum;
#pragma unroll
    for (int d = 0; d < 16; ++d) attn[h][c][d] = row[d] * inv;
  }
  __syncthreads();
  for (int i = tid; i < DIMC * DIMC; i += 256) {
    int o = i >> 7, cd = i & 127, h = cd >> 4, d = cd & 15;
    float s = 0.f;
#pragma unroll
    for (int c = 0; c < 16; ++c)
      s += proj_w[o * DIMC + h * CH + c] * attn[h][c][d];
    Wc[(size_t)b * DIMC * DIMC + i] = f2bf(s);
  }
}

// ---------------------------------------------------------------------------
// K5: out[b,o,n] = sum_cg Wc[b][o][cg] * v[b,cg,n]  (bf16 WMMA 16x16x32).
// Same wave structure as K1; v read once, output written once (f32).
// ---------------------------------------------------------------------------
__global__ __launch_bounds__(256) void k5_out_gemm(const unsigned short* __restrict__ qkvc,
                                                   const unsigned short* __restrict__ Wc,
                                                   float* __restrict__ out) {
  extern __shared__ unsigned short wLds[];             // 128 rows * 130 bf16
  const int b = blockIdx.y;
  const int tid = threadIdx.x;
  for (int i = tid; i < DIMC * DIMC; i += 256) {
    int o = i >> 7, c = i & 127;
    wLds[o * 130 + c] = Wc[(size_t)b * DIMC * DIMC + i];
  }
  __syncthreads();

  const int wave = tid >> 5, lane = tid & 31;
  const int col = lane & 15, half = lane >> 4;
  const size_t sp = ((size_t)blockIdx.x * 8 + wave) * 16;
  const unsigned short* vb = qkvc + ((size_t)b * QKVC + 256) * HW;  // v channels
  const size_t myPix = sp + col;

  Frag16 bfrag[4];
#pragma unroll
  for (int ks2 = 0; ks2 < 4; ++ks2) {
    __builtin_prefetch(vb + (size_t)(32 * ks2 + 16 * half) * HW + myPix, 0, 1);
#pragma unroll
    for (int e = 0; e < 16; ++e) {
      int c = 32 * ks2 + 16 * half + e;
      bfrag[ks2].s[e] = vb[(size_t)c * HW + myPix];
    }
  }

  for (int ot = 0; ot < 8; ++ot) {
    v8f acc = {0.f, 0.f, 0.f, 0.f, 0.f, 0.f, 0.f, 0.f};
#pragma unroll
    for (int ks2 = 0; ks2 < 4; ++ks2) {
      Frag16 af;
      const unsigned short* arow = wLds + (ot * 16 + col) * 130 + 32 * ks2 + (half ? 16 : 0);
#pragma unroll
      for (int v = 0; v < 8; ++v) {
        int c0 = 2 * v;
        af.d[v] = *(const unsigned int*)(arow + c0);
      }
      acc = __builtin_amdgcn_wmma_f32_16x16x32_bf16(false, af.v, false, bfrag[ks2].v,
                                                    (short)0, acc, false, false);
    }
#pragma unroll
    for (int r = 0; r < 8; ++r)
      out[((size_t)b * DIMC + ot * 16 + r + 8 * half) * HW + sp + col] = acc[r];
  }
}

// ---------------------------------------------------------------------------
extern "C" void kernel_launch(void* const* d_in, const int* in_sizes, int n_in,
                              void* d_out, int out_size, void* d_ws, size_t ws_size,
                              hipStream_t stream) {
  (void)in_sizes; (void)n_in; (void)out_size; (void)ws_size;

  const float* x      = (const float*)d_in[0];
  const float* qkv_w  = (const float*)d_in[1];
  const float* temp   = (const float*)d_in[2];
  ConvW cw;
  for (int p = 0; p < 3; ++p) {
    cw.hw_w[p] = (const float*)d_in[3 + p * 6 + 0];
    cw.hw_b[p] = (const float*)d_in[3 + p * 6 + 1];
    cw.w_w[p]  = (const float*)d_in[3 + p * 6 + 2];
    cw.w_b[p]  = (const float*)d_in[3 + p * 6 + 3];
    cw.h_w[p]  = (const float*)d_in[3 + p * 6 + 4];
    cw.h_b[p]  = (const float*)d_in[3 + p * 6 + 5];
  }
  const float* proj_w = (const float*)d_in[21];

  // Workspace carve-up
  unsigned short* qkv  = (unsigned short*)d_ws;
  unsigned short* qkvc = qkv + (size_t)BSZ * QKVC * HW;
  float* sumsq = (float*)(qkvc + (size_t)BSZ * QKVC * HW);
  float* G     = sumsq + BSZ * 256;                     // contiguous with sumsq
  unsigned short* Wc = (unsigned short*)(G + BSZ * HEADS * 256);
  float* out = (float*)d_out;

  const int nZero = BSZ * 256 + BSZ * HEADS * 256;      // sumsq + G
  k0_zero<<<dim3((nZero + 255) / 256), 256, 0, stream>>>(sumsq, nZero);

  k1_qkv_gemm<<<dim3(2048), 256, QKVC * 130 * sizeof(unsigned short), stream>>>(x, qkv_w, qkv);

  k2_inception<<<dim3(HDIM, QKVC, BSZ), 256, 0, stream>>>(qkv, qkvc, sumsq, cw);

  k3_gram<<<dim3(8, BSZ * HEADS), 256,
            4 * K3_TILE_US * sizeof(unsigned short) + 256 * sizeof(float), stream>>>(qkvc, G);

  k4_attn<<<dim3(BSZ), 256, 0, stream>>>(G, sumsq, temp, proj_w, Wc);

  k5_out_gemm<<<dim3(512, BSZ), 256, DIMC * 130 * sizeof(unsigned short), stream>>>(qkvc, Wc, out);
}
